// Encoder_26182120636463
// MI455X (gfx1250) — matrix operand order repaired and checked by reading, not compiled, not run on hardware
//
#include <hip/hip_runtime.h>
#include <hip/hip_bf16.h>
#include <math.h>

typedef __attribute__((ext_vector_type(16))) __bf16 v16bf;
typedef __attribute__((ext_vector_type(8)))  float  v8f;

union Frag {
    v16bf  v;
    uint4  q[2];
    __bf16 e[16];
};

__device__ __forceinline__ unsigned short f2bf_bits(float f) {
    union { float f; unsigned u; } x; x.f = f;
    unsigned r = x.u + 0x7FFFu + ((x.u >> 16) & 1u);   // round-to-nearest-even
    return (unsigned short)(r >> 16);
}
__device__ __forceinline__ __bf16 f2bf(float f) {
    union { unsigned short s; __bf16 b; } y; y.s = f2bf_bits(f);
    return y.b;
}

// Branchless tanh: tanh(|x|) = (1-t)/(1+t) = 1 - 2t/(1+t), t = e^{-2|x|}.
// Uses v_exp_f32 (2^x) and v_rcp_f32; ~1e-6 rel error, far below bf16 quantization.
__device__ __forceinline__ float fast_tanh(float x) {
    float a = __builtin_fabsf(x);
    float t = __builtin_amdgcn_exp2f(a * -2.885390081777927f);  // e^{-2a}
    float r = __builtin_amdgcn_rcpf(1.0f + t);
    float y = 1.0f - (2.0f * t) * r;
    return __builtin_copysignf(y, x);
}

__device__ __forceinline__ void tanh_pack8(const float* __restrict__ p, Frag& a, int eoff) {
    float4 f0 = *(const float4*)(p);
    float4 f1 = *(const float4*)(p + 4);
    a.e[eoff + 0] = f2bf(fast_tanh(f0.x));
    a.e[eoff + 1] = f2bf(fast_tanh(f0.y));
    a.e[eoff + 2] = f2bf(fast_tanh(f0.z));
    a.e[eoff + 3] = f2bf(fast_tanh(f0.w));
    a.e[eoff + 4] = f2bf(fast_tanh(f1.x));
    a.e[eoff + 5] = f2bf(fast_tanh(f1.y));
    a.e[eoff + 6] = f2bf(fast_tanh(f1.z));
    a.e[eoff + 7] = f2bf(fast_tanh(f1.w));
}

// One-time (per call) f32 -> bf16 conversion of W (16*4*128*128 elements).
__global__ void convert_w_kernel(const float* __restrict__ W,
                                 unsigned short* __restrict__ Wbf, int n) {
    int i = blockIdx.x * 256 + threadIdx.x;
    if (i < n) Wbf[i] = f2bf_bits(W[i]);
}

// One block per tree node. 128 threads = 4 waves; wave w owns batch rows 16w..16w+15.
// C strip per wave: 16 x 128, K = 512 (4 children x d=128), via v_wmma_f32_16x16x32_bf16.
// Weight slice W[rule] (4*128*128 bf16 = 128KB) staged through 64KB of LDS in two
// phases (2 children each). LDS is addressed in 16B chunks (uint4) with an XOR swizzle
// on (o&15) so B-fragment ds_load_b128 reads are bank-conflict-free.
template <bool LEAF>
__global__ void __launch_bounds__(128)
tree_level(const int*            __restrict__ rules,       // per-node rule ids (this level)
           const int*            __restrict__ leaf_rules,  // (B, 4096)         [LEAF only]
           const float*          __restrict__ leaf_bias,   // (512, 128)        [LEAF only]
           const unsigned short* __restrict__ h_in,        // (B, 4*n_l, 128) bf16 [!LEAF]
           const unsigned short* __restrict__ Wbf,         // (16,4,128,128) bf16
           const float*          __restrict__ bvec,        // (16, 128)
           unsigned short*       __restrict__ h_out,       // (B, n_l, 128) bf16
           float*                __restrict__ out_f32)     // (B, 128), level 0 only
{
    extern __shared__ uint4 ldsW4[];                       // 4096 chunks = 64 KB

    const int node = blockIdx.x;
    const int n_l  = gridDim.x;
    const int rule = rules[node];

    const int lane = threadIdx.x & 31;
    const int wave = threadIdx.x >> 5;
    const int half = lane >> 4;     // which 16-lane half of the wave
    const int l16  = lane & 15;
    const int b0   = wave * 16;     // batch-row base of this wave's strip

    v8f acc[8];
#pragma unroll
    for (int t = 0; t < 8; ++t)
#pragma unroll
        for (int r = 0; r < 8; ++r) acc[t][r] = 0.0f;

#pragma unroll
    for (int kk2 = 0; kk2 < 2; ++kk2) {                    // two staging phases (children 0,1 / 2,3)
        __syncthreads();                                   // protect previous phase's LDS reads
        {
            const uint4* wsrc4 = (const uint4*)((const unsigned int*)Wbf +
                                                (size_t)rule * 32768u + (size_t)kk2 * 16384u);
            for (int idx = threadIdx.x; idx < 4096; idx += 128) {
                int row = idx >> 4;                        // kloc*128 + o   (0..255)
                int c   = idx & 15;                        // 16B chunk in row
                int o   = row & 127;
                ldsW4[(row << 4) + (c ^ (o & 15))] = wsrc4[idx];
            }
        }
        __syncthreads();

#pragma unroll
        for (int kloc = 0; kloc < 2; ++kloc) {
            const int k = kk2 * 2 + kloc;                  // child index 0..3

            const float*          lbrow = nullptr;
            const unsigned short* hrow  = nullptr;
            if (LEAF) {
                int lr = leaf_rules[(size_t)(b0 + l16) * 4096u + (size_t)node * 4u + k];
                lbrow  = leaf_bias + (size_t)lr * 128u;
            } else {
                hrow = h_in +
                       (((size_t)(b0 + l16) * (size_t)(4 * n_l)) + (size_t)node * 4u + k) * 128u;
            }

#pragma unroll
            for (int si = 0; si < 4; ++si) {               // K window of 32 within this child
                // ---- A fragment (16x32 bf16): per-lane chunks at K = half*8 and half*8+16
                Frag a;
                const int colA = si * 32 + half * 8;
                if (LEAF) {
                    tanh_pack8(lbrow + colA,      a, 0);
                    tanh_pack8(lbrow + colA + 16, a, 8);
                } else {
                    a.q[0] = *(const uint4*)(hrow + colA);
                    a.q[1] = *(const uint4*)(hrow + colA + 16);
                }

                // ---- 8 output tiles: B fragment (32x16 bf16) from LDS, K = half*8 and half*8+16
                const int c0     = si * 4 + half;          // 16B-chunk index within the 256B row
                const int rbase0 = (kloc * 128 + l16) << 4;
#pragma unroll
                for (int t = 0; t < 8; ++t) {
                    Frag bm;
                    const int rbase = rbase0 + (t << 8);   // +t*16 rows = +256 chunks
                    bm.q[0] = ldsW4[rbase + ((c0    ) ^ l16)];
                    bm.q[1] = ldsW4[rbase + ((c0 + 2) ^ l16)];
                    acc[t] = __builtin_amdgcn_wmma_f32_16x16x32_bf16(
                        false, a.v, false, bm.v, (short)0, acc[t], false, false);
                }
            }
        }
    }

    // ---- epilogue: bias + tanh, store (C/D layout: row = half*8+r, col = l16)
#pragma unroll
    for (int t = 0; t < 8; ++t) {
        const int   o  = t * 16 + l16;
        const float bb = bvec[rule * 128 + o];
#pragma unroll
        for (int r = 0; r < 8; ++r) {
            float v   = fast_tanh(acc[t][r] + bb);
            int  brow = b0 + half * 8 + r;
            if (out_f32) {
                out_f32[(size_t)brow * 128u + o] = v;
            } else {
                h_out[((size_t)brow * (size_t)n_l + (size_t)node) * 128u + o] = f2bf_bits(v);
            }
        }
    }
}

extern "C" void kernel_launch(void* const* d_in, const int* in_sizes, int n_in,
                              void* d_out, int out_size, void* d_ws, size_t ws_size,
                              hipStream_t stream) {
    (void)in_sizes; (void)n_in; (void)out_size; (void)ws_size;

    const int*   leaf_rules     = (const int*)d_in[0];    // (64, 4096)
    const int*   internal_rules = (const int*)d_in[1];    // (1365,)
    const float* leaf_bias      = (const float*)d_in[2];  // (512, 128)
    const float* W              = (const float*)d_in[3];  // (16, 4, 128, 128)
    const float* bvec           = (const float*)d_in[4];  // (16, 128)
    float*       out            = (float*)d_out;          // (64, 128)

    unsigned short* Wbf = (unsigned short*)d_ws;                                // 2 MB
    unsigned short* hA  = (unsigned short*)((char*)d_ws + (size_t)(2u  << 20)); // <=16 MB
    unsigned short* hB  = (unsigned short*)((char*)d_ws + (size_t)(18u << 20)); // <=4 MB

    const int NW = 16 * 4 * 128 * 128;
    convert_w_kernel<<<(NW + 255) / 256, 256, 0, stream>>>(W, Wbf, NW);

    const size_t ldsBytes = 4096 * sizeof(uint4);          // 64 KB
    const int offs[6] = {0, 1, 5, 21, 85, 341};            // level -> internal_rules offset

    // level 5: fused leaf gather+tanh -> hA (64, 1024, 128)
    tree_level<true ><<<1024, 128, ldsBytes, stream>>>(internal_rules + offs[5], leaf_rules,
                                                       leaf_bias, nullptr, Wbf, bvec, hA, nullptr);
    // levels 4..1: bf16 ping-pong
    tree_level<false><<< 256, 128, ldsBytes, stream>>>(internal_rules + offs[4], nullptr, nullptr,
                                                       hA, Wbf, bvec, hB, nullptr);
    tree_level<false><<<  64, 128, ldsBytes, stream>>>(internal_rules + offs[3], nullptr, nullptr,
                                                       hB, Wbf, bvec, hA, nullptr);
    tree_level<false><<<  16, 128, ldsBytes, stream>>>(internal_rules + offs[2], nullptr, nullptr,
                                                       hA, Wbf, bvec, hB, nullptr);
    tree_level<false><<<   4, 128, ldsBytes, stream>>>(internal_rules + offs[1], nullptr, nullptr,
                                                       hB, Wbf, bvec, hA, nullptr);
    // level 0: final f32 output to d_out
    tree_level<false><<<   1, 128, ldsBytes, stream>>>(internal_rules + offs[0], nullptr, nullptr,
                                                       hA, Wbf, bvec, nullptr, out);
}
